// GPTLTP_82815559401512
// MI455X (gfx1250) — compile-verified, matrix operand order checked by reading, not run on hardware
//
#include <hip/hip_runtime.h>
#include <hip/hip_bf16.h>
#include <math.h>

// Problem constants (B,T,C,H fixed by the reference)
#define NB 4
#define NT 1024
#define NC 1024
#define NH 16
#define ND 64
#define NSINK 4

typedef __attribute__((ext_vector_type(16))) __bf16 v16bf;
typedef __attribute__((ext_vector_type(8)))  __bf16 v8bf;
typedef __attribute__((ext_vector_type(8)))  float  v8f;

// ---------------------------------------------------------------------------
// WMMA helpers (CDNA5: D = A(16x32) * B(32x16) + C, bf16 in / f32 accum)
// ---------------------------------------------------------------------------
__device__ inline v8f wmma_bf16(v16bf a, v16bf b, v8f c) {
  // 8 args: (neg_a, A, neg_b, B, c_mod, C, reuse_a, reuse_b)
  return __builtin_amdgcn_wmma_f32_16x16x32_bf16(false, a, false, b, (short)0, c,
                                                 false, false);
}

// Load a 16x32 bf16 fragment. Matrix is row-major (ld elements per row).
// ISA layout (16-bit A 16x32): lanes 0-15 hold K = {0..7, 16..23}, lanes
// 16-31 hold K = {8..15, 24..31}; row = lane%16.  The same striping is used
// for B when B is stored transposed (Bt is N x K, row n = column n of B).
__device__ inline v16bf load_frag_bf(const __bf16* base, int ld, int row0, int k0,
                                     int lane) {
  const __bf16* p = base + (size_t)(row0 + (lane & 15)) * ld + k0 + ((lane >> 4) << 3);
  v8bf lo = *reinterpret_cast<const v8bf*>(p);        // K chunk +0..7
  v8bf hi = *reinterpret_cast<const v8bf*>(p + 16);   // K chunk +16..23
  v16bf f;
#pragma unroll
  for (int i = 0; i < 8; ++i) { f[i] = lo[i]; f[i + 8] = hi[i]; }
  return f;
}

// Same fragment but sourced from an fp32 LDS tile (convert on the fly).
__device__ inline v16bf load_frag_lds(const float* S, int ldS, int row0, int k0,
                                      int lane) {
  const float* p = S + (size_t)(row0 + (lane & 15)) * ldS + k0 + ((lane >> 4) << 3);
  v16bf f;
#pragma unroll
  for (int i = 0; i < 8; ++i) {
    f[i]     = (__bf16)p[i];
    f[i + 8] = (__bf16)p[i + 16];
  }
  return f;
}

__device__ inline float wave_sum(float v) {
#pragma unroll
  for (int o = 16; o > 0; o >>= 1) v += __shfl_xor(v, o);
  return v;
}
__device__ inline float wave_max(float v) {
#pragma unroll
  for (int o = 16; o > 0; o >>= 1) v = fmaxf(v, __shfl_xor(v, o));
  return v;
}

// ---------------------------------------------------------------------------
// Weight convert + transpose: W (K x N) fp32 -> Wt (N x K) bf16
// ---------------------------------------------------------------------------
__global__ void gpt_convT(const float* __restrict__ W, __bf16* __restrict__ Out,
                          int K, int N) {
  size_t idx = (size_t)blockIdx.x * 256 + threadIdx.x;  // over N*K, Out-major
  if (idx >= (size_t)N * K) return;
  int n = (int)(idx / K);
  int k = (int)(idx % K);
  Out[idx] = (__bf16)W[(size_t)k * N + n];
}

// ---------------------------------------------------------------------------
// LayerNorm (fp32 in -> bf16 out), one row per workgroup
// ---------------------------------------------------------------------------
__global__ void gpt_layernorm(const float* __restrict__ X, const float* __restrict__ w,
                              const float* __restrict__ b, __bf16* __restrict__ Out,
                              int Cc) {
  __shared__ float red[16];
  int row = blockIdx.x;
  const float* xr = X + (size_t)row * Cc;
  float s = 0.f, s2 = 0.f;
  for (int c = threadIdx.x; c < Cc; c += 256) {
    float v = xr[c];
    s += v; s2 += v * v;
  }
  s = wave_sum(s); s2 = wave_sum(s2);
  int wv = threadIdx.x >> 5, ln = threadIdx.x & 31;
  if (ln == 0) { red[wv] = s; red[8 + wv] = s2; }
  __syncthreads();
  if (threadIdx.x == 0) {
    float a = 0.f, q = 0.f;
#pragma unroll
    for (int i = 0; i < 8; ++i) { a += red[i]; q += red[8 + i]; }
    red[0] = a; red[8] = q;
  }
  __syncthreads();
  float mu = red[0] / Cc;
  float var = red[8] / Cc - mu * mu;
  float rstd = rsqrtf(var + 1e-5f);
  for (int c = threadIdx.x; c < Cc; c += 256)
    Out[(size_t)row * Cc + c] = (__bf16)((xr[c] - mu) * rstd * w[c] + b[c]);
}

// ---------------------------------------------------------------------------
// Generic WMMA GEMM: C(MxN) = A(MxK) * Bt(NxK)^T, block=256 thr (8 waves),
// block tile 128x64, each wave a 32x32 tile (2x2 WMMA tiles, K step 32).
// MODE 1: +bias, scatter to q/k/v (q scaled 1/sqrt(D), v stored transposed)
// MODE 2: +bias +residual -> fp32
// MODE 3: +bias, exact GELU -> bf16
// ---------------------------------------------------------------------------
template <int MODE>
__global__ void gpt_wmma_gemm(const __bf16* __restrict__ A, const __bf16* __restrict__ Bt,
                              int M, int N, int K, const float* __restrict__ bias,
                              const float* __restrict__ resid, float* __restrict__ outF,
                              __bf16* __restrict__ outB, __bf16* __restrict__ outQ,
                              __bf16* __restrict__ outK, __bf16* __restrict__ outV) {
  int tid = threadIdx.x, wave = tid >> 5, lane = tid & 31;
  int wm = wave & 3, wn = wave >> 2;
  int tm = blockIdx.y * 128 + wm * 32;
  int tn = blockIdx.x * 64 + wn * 32;

  v8f acc00 = {}, acc01 = {}, acc10 = {}, acc11 = {};
  for (int k0 = 0; k0 < K; k0 += 32) {
    v16bf a0 = load_frag_bf(A, K, tm, k0, lane);
    v16bf a1 = load_frag_bf(A, K, tm + 16, k0, lane);
    v16bf b0 = load_frag_bf(Bt, K, tn, k0, lane);
    v16bf b1 = load_frag_bf(Bt, K, tn + 16, k0, lane);
    acc00 = wmma_bf16(a0, b0, acc00);
    acc01 = wmma_bf16(a0, b1, acc01);
    acc10 = wmma_bf16(a1, b0, acc10);
    acc11 = wmma_bf16(a1, b1, acc11);
  }

  v8f* accs[4] = {&acc00, &acc01, &acc10, &acc11};
#pragma unroll
  for (int ij = 0; ij < 4; ++ij) {
    int i = ij >> 1, j = ij & 1;
    int row0 = tm + i * 16 + ((lane >> 4) << 3);
    int col = tn + j * 16 + (lane & 15);
    float bcol = bias[col];
#pragma unroll
    for (int e = 0; e < 8; ++e) {
      int row = row0 + e;
      float v = (*accs[ij])[e] + bcol;
      if constexpr (MODE == 1) {
        int sect = col >> 10, cc = col & 1023;
        int hh = cc >> 6, dd = cc & 63;
        int bb = row >> 10, t = row & 1023;
        size_t bht = ((size_t)bb * NH + hh);
        if (sect == 0)
          outQ[(bht * NT + t) * ND + dd] = (__bf16)(v * 0.125f);  // 1/sqrt(64)
        else if (sect == 1)
          outK[(bht * NT + t) * ND + dd] = (__bf16)v;
        else
          outV[(bht * ND + dd) * NT + t] = (__bf16)v;             // transposed
      } else if constexpr (MODE == 2) {
        size_t o = (size_t)row * N + col;
        outF[o] = v + resid[o];
      } else {  // MODE 3: exact GELU
        float g = 0.5f * v * (1.0f + erff(v * 0.70710678118654752f));
        outB[(size_t)row * N + col] = (__bf16)g;
      }
    }
  }
}

// ---------------------------------------------------------------------------
// Fused causal attention for one (b,h) and 32 query rows per workgroup.
// Scores strip (32 x T) lives in LDS (128 KB of the 320 KB/WGP).
// Phase 1: S = q*k^T via WMMA (tiles above the causal diagonal skipped)
// Phase 2: masked softmax per row (wave-parallel) + importance partials
// Phase 3: y = att * v via WMMA (att converted bf16 from LDS)
// ---------------------------------------------------------------------------
__global__ void gpt_attention(const __bf16* __restrict__ hq, const __bf16* __restrict__ hk,
                              const __bf16* __restrict__ hv, const int* __restrict__ amask,
                              __bf16* __restrict__ yb, float* __restrict__ imp) {
  __shared__ float S[32][NT];  // 128 KB
  int bh = blockIdx.x;
  int b = bh >> 4, h = bh & 15;
  int q0 = blockIdx.y * 32;
  int tid = threadIdx.x, wave = tid >> 5, lane = tid & 31;

  const __bf16* Q = hq + (size_t)bh * NT * ND;   // (T, D)
  const __bf16* Kd = hk + (size_t)bh * NT * ND;  // (T, D)
  const __bf16* Vt = hv + (size_t)bh * ND * NT;  // (D, T) transposed

  int kmax = q0 + 32;        // all keys >= kmax are causally masked for this strip
  int ntiles = kmax >> 4;

  // Phase 1: scores (q pre-scaled by 1/sqrt(D) at QKV scatter)
  v16bf aq[2][2];
#pragma unroll
  for (int mt = 0; mt < 2; ++mt)
#pragma unroll
    for (int ks = 0; ks < 2; ++ks)
      aq[mt][ks] = load_frag_bf(Q, ND, q0 + mt * 16, ks * 32, lane);

  for (int nt = wave; nt < ntiles; nt += 8) {
    v16bf bk0 = load_frag_bf(Kd, ND, nt * 16, 0, lane);
    v16bf bk1 = load_frag_bf(Kd, ND, nt * 16, 32, lane);
#pragma unroll
    for (int mt = 0; mt < 2; ++mt) {
      v8f acc = {};
      acc = wmma_bf16(aq[mt][0], bk0, acc);
      acc = wmma_bf16(aq[mt][1], bk1, acc);
      int col = nt * 16 + (lane & 15);
      int rbase = mt * 16 + ((lane >> 4) << 3);
#pragma unroll
      for (int e = 0; e < 8; ++e) S[rbase + e][col] = acc[e];
    }
  }
  __syncthreads();

  // Phase 2: masked softmax, one row per wave (4 rows each)
  for (int r = wave; r < 32; r += 8) {
    int tq = q0 + r;
    float mx = -INFINITY;
    for (int k = lane; k < kmax; k += 32) {
      bool keep = (k <= tq) && (amask[b * NT + k] != 0);
      float s = keep ? S[r][k] : -INFINITY;
      S[r][k] = s;
      mx = fmaxf(mx, s);
    }
    mx = wave_max(mx);
    float sum = 0.f;
    for (int k = lane; k < kmax; k += 32) {
      float s = S[r][k];
      float p = (mx == -INFINITY) ? 0.f : __expf(s - mx);
      S[r][k] = p;
      sum += p;
    }
    sum = wave_sum(sum);
    float inv = (sum > 0.f) ? (1.0f / sum) : 0.f;  // nan_to_num for masked rows
    for (int k = lane; k < kmax; k += 32) S[r][k] *= inv;
  }
  __syncthreads();

  // Phase 2b: importance partials  imp[b,k] += (1/H) * sum_q att[q,k]
  for (int k = tid; k < kmax; k += 256) {
    float a = 0.f;
#pragma unroll 4
    for (int r = 0; r < 32; ++r) a += S[r][k];
    atomicAdd(&imp[b * NT + k], a * (1.0f / (float)NH));
  }
  __syncthreads();

  // Phase 3: y (32 x 64) = att (32 x kmax) @ v (kmax x 64); one tile per wave
  int mt = wave >> 2;  // 0..1
  int dt = wave & 3;   // 0..3
  v8f acc = {};
  for (int k0 = 0; k0 < kmax; k0 += 32) {
    v16bf af = load_frag_lds(&S[0][0], NT, mt * 16, k0, lane);
    v16bf bf = load_frag_bf(Vt, NT, dt * 16, k0, lane);
    acc = wmma_bf16(af, bf, acc);
  }
  int col = h * ND + dt * 16 + (lane & 15);
  int rbase = q0 + mt * 16 + ((lane >> 4) << 3);
#pragma unroll
  for (int e = 0; e < 8; ++e)
    yb[((size_t)b * NT + rbase + e) * NC + col] = (__bf16)acc[e];
}

// ---------------------------------------------------------------------------
// Importance finalization
// ---------------------------------------------------------------------------
__global__ void gpt_init_zero(float* __restrict__ imp, unsigned* __restrict__ gmax) {
  int i = blockIdx.x * 256 + threadIdx.x;
  if (i < NB * NT) imp[i] = 0.f;
  if (i == 0) *gmax = 0u;
}

__global__ void gpt_imp_norm_max(float* __restrict__ imp, unsigned* __restrict__ gmax) {
  __shared__ float red[8];
  int idx = blockIdx.x * 256 + threadIdx.x;
  float v = 0.f;
  if (idx < NB * NT) {
    int t = idx & (NT - 1);
    float pos = (float)(NT - t) * (1.0f / (float)NT);
    v = imp[idx] / (pos + 1e-8f);
    imp[idx] = v;
  }
  float m = wave_max(v);
  int wv = threadIdx.x >> 5, ln = threadIdx.x & 31;
  if (ln == 0) red[wv] = m;
  __syncthreads();
  if (threadIdx.x == 0) {
    float bm = red[0];
#pragma unroll
    for (int i = 1; i < 8; ++i) bm = fmaxf(bm, red[i]);
    atomicMax(gmax, __float_as_uint(bm));  // imp >= 0, bit order == float order
  }
}

__global__ void gpt_imp_finish(const float* __restrict__ imp, const unsigned* __restrict__ gmax,
                               const int* __restrict__ amask, const float* __restrict__ thr,
                               float* __restrict__ outMask, float* __restrict__ outImp) {
  int idx = blockIdx.x * 256 + threadIdx.x;
  if (idx >= NB * NT) return;
  int t = idx & (NT - 1);
  float v = imp[idx];
  if (t < NSINK) v = __uint_as_float(*gmax) + 1.0f;
  float m = (v >= *thr) ? 1.f : 0.f;
  if (t < NSINK) m = 1.f;
  m *= (float)amask[idx];
  outMask[idx] = m;
  outImp[idx] = v;
}

// ---------------------------------------------------------------------------
// Host-side orchestration
// ---------------------------------------------------------------------------
extern "C" void kernel_launch(void* const* d_in, const int* in_sizes, int n_in,
                              void* d_out, int out_size, void* d_ws, size_t ws_size,
                              hipStream_t stream) {
  const float* x      = (const float*)d_in[0];
  const int*   amask  = (const int*)d_in[1];
  const float* W_attn = (const float*)d_in[2];
  const float* b_attn = (const float*)d_in[3];
  const float* W_proj = (const float*)d_in[4];
  const float* b_proj = (const float*)d_in[5];
  const float* ln1w   = (const float*)d_in[6];
  const float* ln1b   = (const float*)d_in[7];
  const float* ln2w   = (const float*)d_in[8];
  const float* ln2b   = (const float*)d_in[9];
  const float* W_fc   = (const float*)d_in[10];
  const float* b_fc   = (const float*)d_in[11];
  const float* W_fc2  = (const float*)d_in[12];
  const float* b_fc2  = (const float*)d_in[13];
  const float* thr    = (const float*)d_in[14];

  char* wsp = (char*)d_ws;
  auto alloc = [&](size_t bytes) -> char* {
    char* p = wsp;
    wsp += (bytes + 255) & ~(size_t)255;
    return p;
  };

  const size_t MT = (size_t)NB * NT;  // 4096 rows
  __bf16* WattnT = (__bf16*)alloc((size_t)3 * NC * NC * 2);
  __bf16* WprojT = (__bf16*)alloc((size_t)NC * NC * 2);
  __bf16* WfcT   = (__bf16*)alloc((size_t)4 * NC * NC * 2);
  __bf16* Wfc2T  = (__bf16*)alloc((size_t)4 * NC * NC * 2);
  __bf16* hln1   = (__bf16*)alloc(MT * NC * 2);
  __bf16* hq     = (__bf16*)alloc(MT * NC * 2);
  __bf16* hk     = (__bf16*)alloc(MT * NC * 2);
  __bf16* hv     = (__bf16*)alloc(MT * NC * 2);
  __bf16* yb     = (__bf16*)alloc(MT * NC * 2);
  float*  x2     = (float*)alloc(MT * NC * 4);
  __bf16* hln2   = (__bf16*)alloc(MT * NC * 2);
  __bf16* gact   = (__bf16*)alloc(MT * (size_t)4 * NC * 2);
  float*  impw   = (float*)alloc(MT * 4);
  unsigned* gmax = (unsigned*)alloc(256);

  float* outMain = (float*)d_out;
  float* outMask = outMain + MT * NC;
  float* outImp  = outMask + MT;

  // 0) zero the importance accumulator + global max
  gpt_init_zero<<<16, 256, 0, stream>>>(impw, gmax);

  // 1) weights -> bf16, transposed (N x K)
  gpt_convT<<<(3 * NC * NC) / 256, 256, 0, stream>>>(W_attn, WattnT, NC, 3 * NC);
  gpt_convT<<<(NC * NC) / 256, 256, 0, stream>>>(W_proj, WprojT, NC, NC);
  gpt_convT<<<(4 * NC * NC) / 256, 256, 0, stream>>>(W_fc, WfcT, NC, 4 * NC);
  gpt_convT<<<(4 * NC * NC) / 256, 256, 0, stream>>>(W_fc2, Wfc2T, 4 * NC, NC);

  // 2) ln1(x) -> bf16
  gpt_layernorm<<<(int)MT, 256, 0, stream>>>(x, ln1w, ln1b, hln1, NC);

  // 3) QKV GEMM (4096 x 3072 x 1024) with q/k/v scatter epilogue
  gpt_wmma_gemm<1><<<dim3(3 * NC / 64, (int)MT / 128), 256, 0, stream>>>(
      hln1, WattnT, (int)MT, 3 * NC, NC, b_attn, nullptr, nullptr, nullptr, hq, hk, hv);

  // 4) fused causal attention + importance partials
  gpt_attention<<<dim3(NB * NH, NT / 32), 256, 0, stream>>>(hq, hk, hv, amask, yb, impw);

  // 5) proj GEMM + bias + residual(x) -> x2 (fp32)
  gpt_wmma_gemm<2><<<dim3(NC / 64, (int)MT / 128), 256, 0, stream>>>(
      yb, WprojT, (int)MT, NC, NC, b_proj, x, x2, nullptr, nullptr, nullptr, nullptr);

  // 6) ln2(x2) -> bf16
  gpt_layernorm<<<(int)MT, 256, 0, stream>>>(x2, ln2w, ln2b, hln2, NC);

  // 7) FC GEMM (4096 x 4096 x 1024) + bias + exact GELU -> bf16
  gpt_wmma_gemm<3><<<dim3(4 * NC / 64, (int)MT / 128), 256, 0, stream>>>(
      hln2, WfcT, (int)MT, 4 * NC, NC, b_fc, nullptr, nullptr, gact, nullptr, nullptr, nullptr);

  // 8) FC2 GEMM (4096 x 1024 x 4096) + bias + residual(x2) -> d_out
  gpt_wmma_gemm<2><<<dim3(NC / 64, (int)MT / 128), 256, 0, stream>>>(
      gact, Wfc2T, (int)MT, NC, 4 * NC, b_fc2, x2, outMain, nullptr, nullptr, nullptr, nullptr);

  // 9) importance: divide by pos_norm, global max, sinks, mask
  gpt_imp_norm_max<<<16, 256, 0, stream>>>(impw, gmax);
  gpt_imp_finish<<<16, 256, 0, stream>>>(impw, gmax, amask, thr, outMask, outImp);
}
